// PointNet_FP_Module_25735444037746
// MI455X (gfx1250) — compile-verified, matrix operand order checked by reading, not run on hardware
//
#include <hip/hip_runtime.h>

// ---------------------------------------------------------------------------
// Types for CDNA5 WMMA (wave32) + TDM descriptors
// ---------------------------------------------------------------------------
typedef __bf16 v16bf  __attribute__((ext_vector_type(16)));
typedef __bf16 bf16x8 __attribute__((ext_vector_type(8)));
typedef float  v8f    __attribute__((ext_vector_type(8)));
typedef unsigned short us8 __attribute__((ext_vector_type(8)));
typedef unsigned int u32x4 __attribute__((ext_vector_type(4)));
typedef int          i32x8 __attribute__((ext_vector_type(8)));
typedef int          i32x4 __attribute__((ext_vector_type(4)));

static __device__ inline unsigned short f2bf(float f) {
    union { float f; unsigned u; } v; v.f = f;
    unsigned u = v.u;
    return (unsigned short)((u + 0x7FFFu + ((u >> 16) & 1u)) >> 16);   // RNE
}

static __device__ inline v16bf cat8(bf16x8 lo, bf16x8 hi) {
    return __builtin_shufflevector(lo, hi, 0,1,2,3,4,5,6,7,8,9,10,11,12,13,14,15);
}

// Problem constants
#define BB   8
#define NN   8192
#define MM   2048
#define C1C  128
#define C2C  256
#define CIN  384

// ---------------------------------------------------------------------------
// 0) f32 -> bf16 conversion (weights)
// ---------------------------------------------------------------------------
__global__ __launch_bounds__(256) void f32_to_bf16(const float* __restrict__ src,
                                                   unsigned short* __restrict__ dst,
                                                   int count) {
    int i = blockIdx.x * 256 + threadIdx.x;
    if (i < count) dst[i] = f2bf(src[i]);
}

// ---------------------------------------------------------------------------
// 1) brute-force 3-NN + inverse-distance weights
// ---------------------------------------------------------------------------
__global__ __launch_bounds__(256) void knn3(const float* __restrict__ xyz1,
                                            const float* __restrict__ xyz2,
                                            int* __restrict__ idxO,
                                            float* __restrict__ wgtO) {
    const int blk = blockIdx.x;
    const int b   = blk >> 5;                     // N/256 = 32 blocks per batch
    const int n   = ((blk & 31) << 8) + threadIdx.x;

    const float qx = xyz1[((long)b * 3 + 0) * NN + n];
    const float qy = xyz1[((long)b * 3 + 1) * NN + n];
    const float qz = xyz1[((long)b * 3 + 2) * NN + n];

    __shared__ float sx[512], sy[512], sz[512];

    float d0 = 3.0e38f, d1 = 3.0e38f, d2 = 3.0e38f;
    int   i0 = 0, i1 = 0, i2 = 0;

    for (int mb = 0; mb < MM; mb += 512) {
        __syncthreads();
        for (int t = threadIdx.x; t < 512; t += 256) {
            sx[t] = xyz2[((long)b * 3 + 0) * MM + mb + t];
            sy[t] = xyz2[((long)b * 3 + 1) * MM + mb + t];
            sz[t] = xyz2[((long)b * 3 + 2) * MM + mb + t];
        }
        __syncthreads();
        #pragma unroll 4
        for (int j = 0; j < 512; ++j) {
            float dx = qx - sx[j], dy = qy - sy[j], dz = qz - sz[j];
            float d  = dx * dx + dy * dy + dz * dz;
            int   m  = mb + j;
            if (d < d2) {
                if (d < d1) {
                    if (d < d0) { d2 = d1; i2 = i1; d1 = d0; i1 = i0; d0 = d; i0 = m; }
                    else        { d2 = d1; i2 = i1; d1 = d;  i1 = m; }
                } else          { d2 = d;  i2 = m; }
            }
        }
    }

    d0 = fmaxf(d0, 1e-10f); d1 = fmaxf(d1, 1e-10f); d2 = fmaxf(d2, 1e-10f);
    float r0 = 1.0f / d0, r1 = 1.0f / d1, r2 = 1.0f / d2;
    float rn = 1.0f / (r0 + r1 + r2);

    long o = ((long)b * NN + n) * 3;
    idxO[o]   = i0; idxO[o+1] = i1; idxO[o+2] = i2;
    wgtO[o]   = r0 * rn; wgtO[o+1] = r1 * rn; wgtO[o+2] = r2 * rn;
}

// ---------------------------------------------------------------------------
// 2) interpolate + concat -> bf16 X (B, 384, N)
// ---------------------------------------------------------------------------
__global__ __launch_bounds__(256) void build_x(const float* __restrict__ p1,
                                               const float* __restrict__ p2,
                                               const int* __restrict__ idx,
                                               const float* __restrict__ wgt,
                                               unsigned short* __restrict__ Xc) {
    long i = (long)blockIdx.x * 256 + threadIdx.x;   // over B*384*N
    int  n = (int)(i & (NN - 1));
    int  c = (int)((i >> 13) % CIN);
    int  b = (int)(i / ((long)CIN * NN));
    float v;
    if (c < C2C) {
        long o = ((long)b * NN + n) * 3;
        const float* row = p2 + ((long)b * C2C + c) * MM;
        v = wgt[o]     * row[idx[o]]
          + wgt[o + 1] * row[idx[o + 1]]
          + wgt[o + 2] * row[idx[o + 2]];
    } else {
        v = p1[((long)b * C1C + (c - C2C)) * NN + n];
    }
    Xc[i] = f2bf(v);
}

// ---------------------------------------------------------------------------
// 3) bf16 WMMA GEMM with TDM weight staging + double-buffered X tiles
//    Y(b,Cout,N) = W(Cout,Cin) * X(b,Cin,N), f32 accumulate.
//    block: 256 threads = 8 waves; block tile 64 rows x 64 cols;
//    wave grid 4x2 -> each wave 16 rows x 32 cols (2 accumulators).
// ---------------------------------------------------------------------------
#define XSTR 40   // X tile LDS col stride (bf16 elems): 16B aligned, conflict-safe
#define PADE 8    // TDM LDS pad: 4 DWORDs (=8 bf16) every 64 DWORDs (=128 bf16)

__global__ __launch_bounds__(256) void gemm_bf16(const unsigned short* __restrict__ Wb,
                                                 const unsigned short* __restrict__ X,
                                                 float* __restrict__ Y,
                                                 int Cout, int Cin) {
    // W strip staged once by TDM, padded row stride = Cin + (Cin/128)*PADE elems
    __shared__ unsigned short sW[64 * 408];          // max Cin=384 -> 408 elems/row
    __shared__ unsigned short sX[2][64 * XSTR];      // [buf][col][k] transposed tiles

    const int tid  = threadIdx.x;
    const int lane = tid & 31;
    const int wid  = tid >> 5;
    const int wr   = wid & 3;       // wave row tile 0..3
    const int wc   = wid >> 2;      // wave col tile 0..1

    const int colbase = blockIdx.x * 64;   // over B*N; 64 | N -> single batch
    const int rowbase = blockIdx.y * 64;   // over Cout
    const int b  = colbase / NN;
    const int n0 = colbase % NN;
    const int wstride = Cin + (Cin >> 7) * PADE;   // 408 (Cin=384) / 272 (Cin=256)

    // ---- TDM: DMA the whole 64 x Cin bf16 weight strip into LDS (wave 0) ----
    if (wid == 0) {
        unsigned long long ga = (unsigned long long)(const void*)(Wb + (size_t)rowbase * Cin);
        unsigned ldsa = (unsigned)(unsigned long long)
            (__attribute__((address_space(3))) unsigned short*)&sW[0];
        u32x4 g0;
        g0[0] = 1u;                                          // count=1, user descriptor
        g0[1] = ldsa;                                        // LDS dest byte address
        g0[2] = (unsigned)(ga & 0xFFFFFFFFull);              // global_addr[31:0]
        g0[3] = (unsigned)((ga >> 32) & 0x1FFFFFFull)        // global_addr[56:32]
              | (2u << 30);                                  // type=2 (image)
        i32x8 g1;
        g1[0] = 0x07510000;       // data_size=2B | pad_en | pad_intv=64DW | pad_amt=4DW
        g1[1] = (int)(((unsigned)Cin  & 0xFFFFu) << 16);     // tensor_dim0[15:0]
        g1[2] = (int)(((unsigned)Cout & 0xFFFFu) << 16);     // tensor_dim1[15:0]
        g1[3] = (int)(((unsigned)Cin  & 0xFFFFu) << 16);     // tile_dim0 = Cin
        g1[4] = 64;                                          // tile_dim1 = 64 rows
        g1[5] = Cin;                                         // tensor_dim0_stride[31:0]
        g1[6] = 0;
        g1[7] = 0;
        i32x4 z4 = {0, 0, 0, 0};
        i32x8 z8 = {0, 0, 0, 0, 0, 0, 0, 0};
        __builtin_amdgcn_tensor_load_to_lds(g0, g1, z4, z4, z8, 0);
        __builtin_amdgcn_s_wait_tensorcnt(0);
    }

    v8f acc0 = {}, acc1 = {};

    // X staging coords: thread -> (k row, 16B segment of 8 columns)
    const int xk   = tid >> 3;          // 0..31
    const int xseg = (tid & 7) * 8;     // 0,8,...,56

    // fragment coords
    const int arow  = wr * 16 + (lane & 15);
    const int khalf = (lane >> 4) * 8;  // 0 or 8
    const int bcol0 = wc * 32 + (lane & 15);
    const int bcol1 = bcol0 + 16;

    const int nk = Cin >> 5;

    // preload tile 0: one global b128 per thread, transpose on LDS store
    us8 xreg = *(const us8*)&X[((long)b * Cin + xk) * NN + n0 + xseg];
    #pragma unroll
    for (int j = 0; j < 8; ++j) sX[0][(xseg + j) * XSTR + xk] = xreg[j];
    __syncthreads();    // also publishes the TDM-staged weight strip

    for (int t = 0; t < nk; ++t) {
        const int cur = t & 1;
        const int kb  = t * 32;

        us8 xnext = xreg;
        if (t + 1 < nk)
            xnext = *(const us8*)&X[((long)b * Cin + kb + 32 + xk) * NN + n0 + xseg];
        if (t + 2 < nk)
            __builtin_prefetch(&X[((long)b * Cin + kb + 64 + xk) * NN + n0 + xseg], 0, 3);

        // A fragment (16x32 bf16) from padded W strip
        const int abase = arow * wstride + kb + ((kb >> 7) * PADE);
        bf16x8 alo = *(const bf16x8*)&sW[abase + khalf];
        bf16x8 ahi = *(const bf16x8*)&sW[abase + 16 + khalf];
        v16bf a = cat8(alo, ahi);

        // B fragments (32x16 bf16), two column tiles
        bf16x8 b0l = *(const bf16x8*)&sX[cur][bcol0 * XSTR + khalf];
        bf16x8 b0h = *(const bf16x8*)&sX[cur][bcol0 * XSTR + 16 + khalf];
        bf16x8 b1l = *(const bf16x8*)&sX[cur][bcol1 * XSTR + khalf];
        bf16x8 b1h = *(const bf16x8*)&sX[cur][bcol1 * XSTR + 16 + khalf];

        acc0 = __builtin_amdgcn_wmma_f32_16x16x32_bf16(false, a, false, cat8(b0l, b0h),
                                                       (short)0, acc0, false, false);
        acc1 = __builtin_amdgcn_wmma_f32_16x16x32_bf16(false, a, false, cat8(b1l, b1h),
                                                       (short)0, acc1, false, false);

        if (t + 1 < nk) {
            #pragma unroll
            for (int j = 0; j < 8; ++j) sX[1 - cur][(xseg + j) * XSTR + xk] = xnext[j];
            __syncthreads();
        }
    }

    // epilogue: lanes 0-15 -> M=r, lanes 16-31 -> M=r+8; N = lane&15
    const int mbase = rowbase + wr * 16 + (lane >> 4) * 8;
    const int ncol  = lane & 15;
    #pragma unroll
    for (int r = 0; r < 8; ++r) {
        int row = mbase + r;
        Y[((long)b * Cout + row) * NN + n0 + wc * 32 + ncol]      = acc0[r];
        Y[((long)b * Cout + row) * NN + n0 + wc * 32 + 16 + ncol] = acc1[r];
    }
}

// ---------------------------------------------------------------------------
// 4) BN batch statistics -> per-channel scale/shift
// ---------------------------------------------------------------------------
__global__ __launch_bounds__(256) void bn_stats(const float* __restrict__ Y, int Cout,
                                                const float* __restrict__ g,
                                                const float* __restrict__ bta,
                                                float* __restrict__ scale,
                                                float* __restrict__ shift) {
    const int  c  = blockIdx.x;
    const long BN = (long)BB * NN;
    float s = 0.0f, ss = 0.0f;
    for (long i = threadIdx.x; i < BN; i += 256) {
        long bi = i >> 13;
        long n  = i & (NN - 1);
        float y = Y[(bi * Cout + c) * NN + n];
        s += y; ss += y * y;
    }
    __shared__ float rs[256], rss[256];
    rs[threadIdx.x] = s; rss[threadIdx.x] = ss;
    __syncthreads();
    for (int o = 128; o > 0; o >>= 1) {
        if (threadIdx.x < o) {
            rs[threadIdx.x]  += rs[threadIdx.x + o];
            rss[threadIdx.x] += rss[threadIdx.x + o];
        }
        __syncthreads();
    }
    if (threadIdx.x == 0) {
        float mean = rs[0] / (float)BN;
        float var  = rss[0] / (float)BN - mean * mean;
        float sc   = g[c] * rsqrtf(var + 1e-5f);
        scale[c] = sc;
        shift[c] = bta[c] - mean * sc;
    }
}

// ---------------------------------------------------------------------------
// 5) normalize + LeakyReLU(0.2); bf16 out for hidden layers, f32 for final
// ---------------------------------------------------------------------------
template <bool TO_BF16>
__global__ __launch_bounds__(256) void bn_act(const float* __restrict__ Y,
                                              const float* __restrict__ scale,
                                              const float* __restrict__ shift,
                                              int Cout,
                                              unsigned short* __restrict__ obf,
                                              float* __restrict__ of32) {
    long i = (long)blockIdx.x * 256 + threadIdx.x;
    int  c = (int)((i >> 13) % Cout);
    float v = Y[i] * scale[c] + shift[c];
    v = (v >= 0.0f) ? v : 0.2f * v;
    if (TO_BF16) obf[i] = f2bf(v);
    else         of32[i] = v;
}

// ---------------------------------------------------------------------------
// Launch
// ---------------------------------------------------------------------------
extern "C" void kernel_launch(void* const* d_in, const int* in_sizes, int n_in,
                              void* d_out, int out_size, void* d_ws, size_t ws_size,
                              hipStream_t stream) {
    (void)in_sizes; (void)n_in; (void)out_size; (void)ws_size;
    const float* xyz1    = (const float*)d_in[0];
    const float* xyz2    = (const float*)d_in[1];
    const float* points1 = (const float*)d_in[2];
    const float* points2 = (const float*)d_in[3];
    const float* W0 = (const float*)d_in[4];
    const float* g0 = (const float*)d_in[5];
    const float* b0 = (const float*)d_in[6];
    const float* W1 = (const float*)d_in[7];
    const float* g1 = (const float*)d_in[8];
    const float* b1 = (const float*)d_in[9];
    const float* W2 = (const float*)d_in[10];
    const float* g2 = (const float*)d_in[11];
    const float* b2 = (const float*)d_in[12];

    char* ws = (char*)d_ws;
    unsigned short* Wb0  = (unsigned short*)(ws + 0);          // 256*384*2
    unsigned short* Wb1  = (unsigned short*)(ws + 196608);     // 256*256*2
    unsigned short* Wb2  = (unsigned short*)(ws + 327680);     // 128*256*2
    float*          scl  = (float*)(ws + 393216);              // 256 f32
    float*          shf  = scl + 256;                          // 256 f32
    int*            idx  = (int*)(ws + 396288);                // B*N*3
    float*          wgt  = (float*)(ws + 1183744);             // B*N*3
    unsigned short* Xc   = (unsigned short*)(ws + 2097152);    // B*384*N bf16 (48MB)
    float*          Yb   = (float*)(ws + 52428800);            // B*256*N f32 (64MB)
    unsigned short* Hb   = (unsigned short*)(ws + 119537664);  // B*256*N bf16 (32MB)
    unsigned short* Hb2  = Xc;                                 // reuse X region
    float*          out  = (float*)d_out;                      // B*128*N f32

    // weights -> bf16
    f32_to_bf16<<<(256*384)/256, 256, 0, stream>>>(W0, Wb0, 256*384);
    f32_to_bf16<<<(256*256)/256, 256, 0, stream>>>(W1, Wb1, 256*256);
    f32_to_bf16<<<(128*256)/256, 256, 0, stream>>>(W2, Wb2, 128*256);

    // 3-NN + interpolation + concat
    knn3<<<BB * (NN/256), 256, 0, stream>>>(xyz1, xyz2, idx, wgt);
    build_x<<<(int)(((long)BB*CIN*NN)/256), 256, 0, stream>>>(points1, points2, idx, wgt, Xc);

    const int COLS = BB * NN;   // 65536

    // layer 0: 384 -> 256
    gemm_bf16<<<dim3(COLS/64, 256/64), 256, 0, stream>>>(Wb0, Xc, Yb, 256, CIN);
    bn_stats<<<256, 256, 0, stream>>>(Yb, 256, g0, b0, scl, shf);
    bn_act<true><<<(int)(((long)BB*256*NN)/256), 256, 0, stream>>>(Yb, scl, shf, 256, Hb, nullptr);

    // layer 1: 256 -> 256
    gemm_bf16<<<dim3(COLS/64, 256/64), 256, 0, stream>>>(Wb1, Hb, Yb, 256, 256);
    bn_stats<<<256, 256, 0, stream>>>(Yb, 256, g1, b1, scl, shf);
    bn_act<true><<<(int)(((long)BB*256*NN)/256), 256, 0, stream>>>(Yb, scl, shf, 256, Hb2, nullptr);

    // layer 2: 256 -> 128, final output f32
    gemm_bf16<<<dim3(COLS/64, 128/64), 256, 0, stream>>>(Wb2, Hb2, Yb, 128, 256);
    bn_stats<<<128, 256, 0, stream>>>(Yb, 128, g2, b2, scl, shf);
    bn_act<false><<<(int)(((long)BB*128*NN)/256), 256, 0, stream>>>(Yb, scl, shf, 128, nullptr, out);
}
